// Attention_63282048139700
// MI455X (gfx1250) — matrix operand order repaired and checked by reading, not compiled, not run on hardware
//
#include <hip/hip_runtime.h>
#include <hip/hip_bf16.h>
#include <math.h>

typedef __attribute__((ext_vector_type(16))) __bf16 v16bf;
typedef __attribute__((ext_vector_type(8)))  float  v8f;

// Problem constants (from reference)
constexpr int cB = 2, cS = 1024, cD = 4096, cH = 32, cKVH = 8, cHD = 128;
constexpr int cM = cB * cS;          // 2048 rows in all GEMMs

// ---------------------------------------------------------------------------
// GEMM: C[M,N] = A[M,K] (f32) * W[K,N] (f32), bf16 WMMA, f32 accumulate.
// Block = 256 threads (8 waves). Block tile 64(M) x 64(N), K-tile 64.
// Wave w: rows (w>>1)*16..+15, cols (w&1)*32..+31 (two 16x16 C tiles, shared A frag).
// WMMA operand striping (CDNA5, wave32):
//   A frag (16x32 bf16): lane holds row (lane&15); elem j -> k = 16*half+j
//   B frag (32x16 bf16): lane holds col (lane&15); elem j -> k = 16*half+j
//   C/D  (16x16 f32):    lane, vgpr r -> row r+8*half, col lane&15
// ---------------------------------------------------------------------------
__global__ __launch_bounds__(256) void gemm_bf16(const float* __restrict__ A,
                                                 const float* __restrict__ W,
                                                 float* __restrict__ C,
                                                 int M, int N, int K)
{
    __shared__ __bf16 As[64][72];   // [m][k] (+pad)
    __shared__ __bf16 Bs[64][72];   // [n][k] (transposed so lane reads are contiguous)

    const int tid  = threadIdx.x;
    const int lane = tid & 31;
    const int wave = tid >> 5;
    const int half = lane >> 4;
    const int l16  = lane & 15;

    const int m0 = blockIdx.y * 64;
    const int n0 = blockIdx.x * 64;
    const int rowOff = (wave >> 1) * 16;   // 0,16,32,48
    const int colOff = (wave & 1) * 32;    // 0,32

    // Static staging assignments (fully unrolled, vector-width friendly):
    // A: thread owns row (tid>>2), 16 contiguous cols at ((tid&3)*16)
    const int ra = tid >> 2;
    const int ca = (tid & 3) * 16;
    // W: thread owns a 4(k) x 4(n) micro-tile; transpose in registers
    const int bn = (tid & 15) * 4;
    const int bk = (tid >> 4) * 4;

    const float* srcA = A + (size_t)(m0 + ra) * K + ca;
    const float* srcW = W + (size_t)bk * N + n0 + bn;

    v8f acc0 = {}, acc1 = {};

    for (int kk = 0; kk < K; kk += 64) {
        // ---- stage A: 4x float4 loads, 16 bf16 contiguous stores ----
        {
            float v[16];
            #pragma unroll
            for (int j = 0; j < 16; j += 4) {
                const float4 t = *(const float4*)(srcA + kk + j);
                v[j] = t.x; v[j+1] = t.y; v[j+2] = t.z; v[j+3] = t.w;
            }
            #pragma unroll
            for (int j = 0; j < 16; ++j)
                As[ra][ca + j] = (__bf16)v[j];
        }
        // ---- stage W transposed: 4x float4 loads, 4x4 register transpose ----
        {
            float w[4][4];
            #pragma unroll
            for (int j = 0; j < 4; ++j) {
                const float4 t = *(const float4*)(srcW + (size_t)(kk + j) * N);
                w[j][0] = t.x; w[j][1] = t.y; w[j][2] = t.z; w[j][3] = t.w;
            }
            #pragma unroll
            for (int i = 0; i < 4; ++i)
                #pragma unroll
                for (int j = 0; j < 4; ++j)
                    Bs[bn + i][bk + j] = (__bf16)w[j][i];
        }
        // ---- speculative prefetch of next K-tile (global_prefetch_b8) ----
        if (kk + 64 < K) {
            __builtin_prefetch(srcA + kk + 64, 0, 3);
            __builtin_prefetch(srcW + (size_t)(kk + 64) * N, 0, 3);
        }
        __syncthreads();

        const int arow = rowOff + l16;
        const int bc0  = colOff + l16;
        const int bc1  = colOff + 16 + l16;
        #pragma unroll
        for (int c2 = 0; c2 < 64; c2 += 32) {
            v16bf a, b0, b1;
            #pragma unroll
            for (int j = 0; j < 8; ++j) {
                a[j]     = As[arow][c2 + 16 * half + j];
                a[j + 8] = As[arow][c2 + 16 * half + 8 + j];
            }
            #pragma unroll
            for (int j = 0; j < 16; ++j) {
                b0[j] = Bs[bc0][c2 + 16 * half + j];
                b1[j] = Bs[bc1][c2 + 16 * half + j];
            }
            acc0 = __builtin_amdgcn_wmma_f32_16x16x32_bf16(
                false, a, false, b0, (short)0, acc0, false, false);
            acc1 = __builtin_amdgcn_wmma_f32_16x16x32_bf16(
                false, a, false, b1, (short)0, acc1, false, false);
        }
        __syncthreads();
    }

    // ---- writeback: one row pointer, constant stride ----
    float* crow = C + (size_t)(m0 + rowOff + 8 * half) * N + n0 + colOff + l16;
    #pragma unroll
    for (int r = 0; r < 8; ++r) {
        crow[0]  = acc0[r];
        crow[16] = acc1[r];
        crow += N;
    }
}

// ---------------------------------------------------------------------------
// RoPE (f32 in -> bf16 out): t (B,S,NH,HD), pairs (2i,2i+1) rotated by ang[s,i]
// ---------------------------------------------------------------------------
__global__ void rope_bf16_kernel(const float* __restrict__ src,
                                 __bf16* __restrict__ dst,
                                 const float* __restrict__ cosp,
                                 const float* __restrict__ sinp,
                                 int nh, size_t total_pairs)
{
    size_t idx = (size_t)blockIdx.x * blockDim.x + threadIdx.x;
    if (idx >= total_pairs) return;
    const int HD2 = cHD / 2;
    int i       = (int)(idx % HD2);
    size_t rest = idx / HD2;              // (b*S + s)*nh + h
    int s       = (int)((rest / nh) % cS);
    float c  = cosp[(size_t)s * HD2 + i];
    float sn = sinp[(size_t)s * HD2 + i];
    size_t base = rest * cHD + 2 * (size_t)i;
    float x0 = src[base], x1 = src[base + 1];
    dst[base]     = (__bf16)(x0 * c - x1 * sn);
    dst[base + 1] = (__bf16)(x0 * sn + x1 * c);
}

// ---------------------------------------------------------------------------
// V transpose + downconvert: Vf32 (B,S,KVH,HD) -> Vt bf16 (B,KVH,HD,S)
// ---------------------------------------------------------------------------
__global__ void v_transpose_kernel(const float* __restrict__ V,
                                   __bf16* __restrict__ Vt)
{
    size_t idx = (size_t)blockIdx.x * blockDim.x + threadIdx.x;
    const size_t total = (size_t)cB * cS * cKVH * cHD;
    if (idx >= total) return;
    int d        = (int)(idx % cHD);
    size_t rest  = idx / cHD;
    int kvh      = (int)(rest % cKVH);
    size_t rest2 = rest / cKVH;
    int s        = (int)(rest2 % cS);
    int b        = (int)(rest2 / cS);
    Vt[(((size_t)b * cKVH + kvh) * cHD + d) * cS + s] = (__bf16)V[idx];
}

// ---------------------------------------------------------------------------
// Flash attention (causal, GQA). One wave per 16-query block.
// grid = (S/64, H, B), block = 128 (4 waves, independent q blocks).
// Q bf16 (B,S,H,HD), K bf16 (B,S,KVH,HD), Vt bf16 (B,KVH,HD,S). O f32 (B,S,H,HD).
// ---------------------------------------------------------------------------
__global__ __launch_bounds__(128) void flash_attn(const __bf16* __restrict__ Q,
                                                  const __bf16* __restrict__ K,
                                                  const __bf16* __restrict__ Vt,
                                                  float* __restrict__ O)
{
    __shared__ __bf16 Pst[4][16][32];     // per-wave P tile (16 q x 32 keys)

    const int lane = threadIdx.x & 31;
    const int wave = threadIdx.x >> 5;
    const int half = lane >> 4;
    const int l16  = lane & 15;

    const int q0 = blockIdx.x * 64 + wave * 16;
    const int h  = blockIdx.y;
    const int b  = blockIdx.z;
    const int kvh = h / (cH / cKVH);      // GQA: 4 query heads per kv head

    const __bf16* Qb = Q + ((size_t)b * cS * cH + h) * cHD;
    const __bf16* Kb = K + ((size_t)b * cS * cKVH + kvh) * cHD;
    // Vt row for HD-column d: contiguous in key index
    const __bf16* Vtb = Vt + ((size_t)b * cKVH + kvh) * cHD * cS;

    // Load Q fragments (A-layout), 4 chunks of k=32 covering HD=128
    v16bf qf[4];
    {
        const __bf16* qrow = Qb + (size_t)(q0 + l16) * cH * cHD;
        #pragma unroll
        for (int c = 0; c < 4; ++c) {
            const int kk = 32 * c;
            #pragma unroll
            for (int j = 0; j < 8; ++j) {
                qf[c][j]     = qrow[kk + 16 * half + j];
                qf[c][j + 8] = qrow[kk + 16 * half + 8 + j];
            }
        }
    }

    float mrow[8], lrow[8];
    v8f oacc[8] = {};
    #pragma unroll
    for (int r = 0; r < 8; ++r) { mrow[r] = -1e30f; lrow[r] = 0.0f; }

    const float scale = 0.08838834764831845f;   // 1/sqrt(128)
    const int kbEnd = q0 + 15;                  // causal: keys <= query

    for (int kb = 0; kb <= kbEnd; kb += 32) {
        // ---- scores for two 16-key subtiles: S = Q K^T ----
        v8f s01[2];
        #pragma unroll
        for (int t = 0; t < 2; ++t) {
            const int kt = kb + 16 * t;
            const __bf16* krow = Kb + (size_t)(kt + l16) * cKVH * cHD;
            v8f sc = {};
            #pragma unroll
            for (int c = 0; c < 4; ++c) {
                // B-frag: 16 contiguous bf16 per lane (32B -> 2x b128)
                const v16bf kf = *(const v16bf*)(krow + 32 * c + 16 * half);
                sc = __builtin_amdgcn_wmma_f32_16x16x32_bf16(
                    false, qf[c], false, kf, (short)0, sc, false, false);
            }
            s01[t] = sc;
        }

        // ---- scale + causal mask + online softmax update ----
        float p0[8], p1[8], alpha[8];
        #pragma unroll
        for (int r = 0; r < 8; ++r) {
            const int qidx = q0 + r + 8 * half;
            float v0 = s01[0][r] * scale + ((kb      + l16) > qidx ? -1e30f : 0.0f);
            float v1 = s01[1][r] * scale + ((kb + 16 + l16) > qidx ? -1e30f : 0.0f);
            float vmax = fmaxf(v0, v1);
            #pragma unroll
            for (int mk = 1; mk < 16; mk <<= 1)
                vmax = fmaxf(vmax, __shfl_xor(vmax, mk, 32));
            float nm = fmaxf(mrow[r], vmax);
            alpha[r] = __expf(mrow[r] - nm);
            mrow[r] = nm;
            p0[r] = __expf(v0 - nm);
            p1[r] = __expf(v1 - nm);
            float rsum = p0[r] + p1[r];
            #pragma unroll
            for (int mk = 1; mk < 16; mk <<= 1)
                rsum += __shfl_xor(rsum, mk, 32);
            lrow[r] = lrow[r] * alpha[r] + rsum;
        }

        // ---- rescale running output ----
        #pragma unroll
        for (int t8 = 0; t8 < 8; ++t8)
            #pragma unroll
            for (int r = 0; r < 8; ++r)
                oacc[t8][r] *= alpha[r];

        // ---- restripe P (C-layout -> A-layout) through LDS ----
        #pragma unroll
        for (int r = 0; r < 8; ++r) {
            Pst[wave][r + 8 * half][l16]      = (__bf16)p0[r];
            Pst[wave][r + 8 * half][16 + l16] = (__bf16)p1[r];
        }
        v16bf pa;   // A-frag: 16 queries x 32 keys
        #pragma unroll
        for (int j = 0; j < 8; ++j) {
            pa[j]     = Pst[wave][l16][16 * half + j];
            pa[j + 8] = Pst[wave][l16][16 * half + 8 + j];
        }

        // ---- O += P * V  (8 HD column tiles of 16) ----
        #pragma unroll
        for (int t8 = 0; t8 < 8; ++t8) {
            const int d = 16 * t8 + l16;
            // B-frag from Vt: 16 contiguous keys per lane (32B -> 2x b128)
            const v16bf vf = *(const v16bf*)(Vtb + (size_t)d * cS + kb + 16 * half);
            oacc[t8] = __builtin_amdgcn_wmma_f32_16x16x32_bf16(
                false, pa, false, vf, (short)0, oacc[t8], false, false);
        }
    }

    // ---- normalize and write (B,S,H,HD) ----
    #pragma unroll
    for (int t8 = 0; t8 < 8; ++t8) {
        const int d = 16 * t8 + l16;
        #pragma unroll
        for (int r = 0; r < 8; ++r) {
            const int q = q0 + r + 8 * half;
            O[(((size_t)b * cS + q) * cH + h) * cHD + d] = oacc[t8][r] / lrow[r];
        }
    }
}

// ---------------------------------------------------------------------------
// Host launcher
// inputs: x, start_pos, cos, sin, mask, wq, wk, wv, wo
// ---------------------------------------------------------------------------
extern "C" void kernel_launch(void* const* d_in, const int* in_sizes, int n_in,
                              void* d_out, int out_size, void* d_ws, size_t ws_size,
                              hipStream_t stream)
{
    const float* x    = (const float*)d_in[0];
    const float* cosp = (const float*)d_in[2];
    const float* sinp = (const float*)d_in[3];
    const float* wq   = (const float*)d_in[5];
    const float* wk   = (const float*)d_in[6];
    const float* wv   = (const float*)d_in[7];
    const float* wo   = (const float*)d_in[8];
    float* out = (float*)d_out;

    constexpr size_t nQ  = (size_t)cM * (cH   * cHD);   // 2048*4096
    constexpr size_t nKV = (size_t)cM * (cKVH * cHD);   // 2048*1024

    float* Qb = (float*)d_ws;           // f32 Q  (pre-RoPE)
    float* Kb = Qb + nQ;                // f32 K  (pre-RoPE)
    float* Vb = Kb + nKV;               // f32 V
    float* AO = Vb + nKV;               // f32 attention output
    __bf16* Qbf = (__bf16*)(AO + nQ);   // bf16 Q (post-RoPE)
    __bf16* Kbf = Qbf + nQ;             // bf16 K (post-RoPE)
    __bf16* Vt  = Kbf + nKV;            // bf16 V^T (B,KVH,HD,S)

    // QKV projections (bf16 WMMA GEMMs)
    gemm_bf16<<<dim3((cH   * cHD) / 64, cM / 64), 256, 0, stream>>>(x, wq, Qb, cM, cH   * cHD, cD);
    gemm_bf16<<<dim3((cKVH * cHD) / 64, cM / 64), 256, 0, stream>>>(x, wk, Kb, cM, cKVH * cHD, cD);
    gemm_bf16<<<dim3((cKVH * cHD) / 64, cM / 64), 256, 0, stream>>>(x, wv, Vb, cM, cKVH * cHD, cD);

    // RoPE (f32 -> bf16) and V transpose/downconvert
    {
        size_t qpairs = (size_t)cB * cS * cH   * (cHD / 2);
        size_t kpairs = (size_t)cB * cS * cKVH * (cHD / 2);
        rope_bf16_kernel<<<(unsigned)((qpairs + 255) / 256), 256, 0, stream>>>(Qb, Qbf, cosp, sinp, cH,   qpairs);
        rope_bf16_kernel<<<(unsigned)((kpairs + 255) / 256), 256, 0, stream>>>(Kb, Kbf, cosp, sinp, cKVH, kpairs);
        size_t vtot = (size_t)cB * cS * cKVH * cHD;
        v_transpose_kernel<<<(unsigned)((vtot + 255) / 256), 256, 0, stream>>>(Vb, Vt);
    }

    // Causal flash attention with GQA
    flash_attn<<<dim3(cS / 64, cH, cB), 128, 0, stream>>>(Qbf, Kbf, Vt, AO);

    // Output projection
    gemm_bf16<<<dim3(cD / 64, cM / 64), 256, 0, stream>>>(AO, wo, out, cM, cD, cD);
}
